// Structure_Aware_Pre_Selected_Rendering_25580825215036
// MI455X (gfx1250) — matrix operand order repaired and checked by reading, not compiled, not run on hardware
//
#include <hip/hip_runtime.h>
#include <hip/hip_bf16.h>

#define DEV __device__ __forceinline__

typedef __attribute__((ext_vector_type(16))) _Float16 v16h;
typedef __attribute__((ext_vector_type(8)))  _Float16 v8h;
typedef __attribute__((ext_vector_type(8)))  float    v8f;
typedef __attribute__((ext_vector_type(4)))  unsigned int u32x4;
typedef __attribute__((ext_vector_type(8)))  int          i32x8;
typedef __attribute__((ext_vector_type(4)))  int          i32x4;

constexpr int BB = 4, CC = 64, UU = 5, VV = 5, HH = 64, WW = 64, KT = 9;
constexpr int CIN1 = 96;          // 64 img + 2 pos channels, padded to multiple of 32
constexpr int KDEF = KT * CC;     // 576 = deform-GEMM K dimension
constexpr int BLOCK = 128;        // 4 waves: one 64-px image row per block

// ---------------- workspace layout (bytes, 256-aligned) ----------------
constexpr size_t AL(size_t x) { return (x + 255) & ~size_t(255); }
constexpr size_t WS_W1   = 0;
constexpr size_t WS_W2   = WS_W1   + AL((size_t)KT * CC * CIN1 * 2);
constexpr size_t WS_W3   = WS_W2   + AL((size_t)KT * CC * CC   * 2);
constexpr size_t WS_WF   = WS_W3   + AL((size_t)KT * CC * CC   * 2);
constexpr size_t WS_WD   = WS_WF   + AL((size_t)KT * 16 * CC   * 2);
constexpr size_t WS_AIN  = WS_WD   + AL((size_t)CC * KDEF      * 2);
constexpr size_t WS_AH1  = WS_AIN  + AL((size_t)BB * HH * WW * CIN1 * 2);
constexpr size_t WS_AH2  = WS_AH1  + AL((size_t)BB * HH * WW * CC   * 2);
constexpr size_t WS_AH3  = WS_AH2  + AL((size_t)BB * HH * WW * CC   * 2);
constexpr size_t WS_SCF  = WS_AH3  + AL((size_t)BB * HH * WW * CC   * 2);
constexpr size_t WS_SAMP = WS_SCF  + AL((size_t)BB * HH * WW * 16   * 4);

// ---------------- TDM availability (clang-23 six-arg form) ----------------
#if defined(__gfx1250__) && __has_builtin(__builtin_amdgcn_tensor_load_to_lds) && \
    __has_builtin(__builtin_amdgcn_s_wait_tensorcnt)
#define USE_TDM 1
#else
#define USE_TDM 0
#endif

#if USE_TDM
// 1-D contiguous DMA: ndwords 4-byte elements from gptr -> LDS byte address lds_addr.
// D# per ISA 8.3/8.4: group0 = {count=1 | lds_addr | global_addr | type=2},
// group1 = {data_size=4B, tensor_dim0=tile_dim0=stride=ndwords, tensor_dim1=1}.
// Groups 2/3 (and the trailing i32x8) zero-filled: tensor rank <= 2.
DEV void tdm_load_1d(unsigned lds_addr, const void* gptr, unsigned ndwords)
{
    unsigned long long ga = (unsigned long long)(size_t)gptr;
    u32x4 g0;
    g0[0] = 1u;                                            // count=1 (valid user D#)
    g0[1] = lds_addr;                                      // LDS byte address
    g0[2] = (unsigned)(ga & 0xffffffffu);                  // global_addr[31:0]
    g0[3] = (unsigned)((ga >> 32) & 0x1ffffffu) | (2u << 30); // global_addr[56:32] | type=2
    i32x8 g1;
    g1[0] = (int)(2u << 16);                               // wg_mask=0, data_size=2 (4 bytes)
    g1[1] = (int)((ndwords & 0xffffu) << 16);              // tensor_dim0[15:0]
    g1[2] = (int)(((ndwords >> 16) & 0xffffu) | (1u << 16)); // tensor_dim0[31:16], tensor_dim1=1
    g1[3] = (int)((ndwords & 0xffffu) << 16);              // tile_dim0 = ndwords
    g1[4] = 0;                                             // tile_dim1=0 (unused), tile_dim2=0
    g1[5] = (int)ndwords;                                  // tensor_dim0_stride[31:0]
    g1[6] = 0;                                             // stride[47:32]=0, dim1_stride lo=0
    g1[7] = 0;
    i32x4 z4 = {0, 0, 0, 0};
    i32x8 z8 = {0, 0, 0, 0, 0, 0, 0, 0};
    __builtin_amdgcn_tensor_load_to_lds(g0, g1, z4, z4, z8, 0);
}
#endif

// ---------------- weight repack: fp32 (O,I,3,3) -> f16 [tap][oc][ic] ----------------
__global__ void prep_weights(const float* __restrict__ w1, const float* __restrict__ w2,
                             const float* __restrict__ w3, const float* __restrict__ wf,
                             const float* __restrict__ wd,
                             _Float16* __restrict__ g1, _Float16* __restrict__ g2,
                             _Float16* __restrict__ g3, _Float16* __restrict__ gf,
                             _Float16* __restrict__ gd)
{
    int idx = blockIdx.x * blockDim.x + threadIdx.x;
    if (idx < KT * CC * CIN1) {                 // conv1: Cin 66 padded to 96
        int t = idx / (CC * CIN1); int r = idx % (CC * CIN1);
        int o = r / CIN1; int ci = r % CIN1;
        g1[idx] = (ci < 66) ? (_Float16)w1[(o * 66 + ci) * 9 + t] : (_Float16)0.f;
    }
    if (idx < KT * CC * CC) {                   // conv2 / conv3
        int t = idx / (CC * CC); int r = idx % (CC * CC);
        int o = r / CC; int ci = r % CC;
        g2[idx] = (_Float16)w2[(o * CC + ci) * 9 + t];
        g3[idx] = (_Float16)w3[(o * CC + ci) * 9 + t];
    }
    if (idx < KT * 16 * CC) {                   // score conv: 9 oc padded to 16
        int t = idx / (16 * CC); int r = idx % (16 * CC);
        int o = r / CC; int ci = r % CC;
        gf[idx] = (o < 9) ? (_Float16)wf[(o * CC + ci) * 9 + t] : (_Float16)0.f;
    }
    if (idx < CC * KDEF) {                      // deform: [t][o][c] = wd[o][c][t]
        int t = idx / (CC * CC); int r = idx % (CC * CC);
        int o = r / CC; int c = r % CC;
        gd[idx] = (_Float16)wd[(o * CC + c) * 9 + t];
    }
}

// ---------------- pack NCHW(UV) slice + pos channels -> NHWC f16 [b][y][x][96] ----------------
__global__ void pack_input(const float* __restrict__ x, const float* __restrict__ vu,
                           const float* __restrict__ vv, const float* __restrict__ mg,
                           _Float16* __restrict__ ain, int vi, int vj)
{
    int p = blockIdx.x * blockDim.x + threadIdx.x;
    if (p >= BB * HH * WW) return;
    int b = p / (HH * WW); int yx = p % (HH * WW);
    float denom = mg[UU - 1] - mg[0];
    float pdu = (vu[b] - mg[vi]) / denom;
    float pdv = (vv[b] - mg[vj]) / denom;
    const float* src = x + (((size_t)b * CC * UU + vi) * VV + vj) * (HH * WW) + yx;
    _Float16* dst = ain + (size_t)p * CIN1;
#pragma unroll
    for (int c = 0; c < CC; ++c)
        dst[c] = (_Float16)src[(size_t)c * UU * VV * HH * WW];
    dst[64] = (_Float16)pdu;
    dst[65] = (_Float16)pdv;
#pragma unroll
    for (int c = 66; c < CIN1; ++c) dst[c] = (_Float16)0.f;
}

// ---------------- bilinear gather (PNX/PNY cancel: py = y + sc*pu, px = x + sc*pv) ----------------
__global__ void gather_samples(const float* __restrict__ x, const float* __restrict__ scf,
                               const float* __restrict__ vu, const float* __restrict__ vv,
                               const float* __restrict__ mg, _Float16* __restrict__ samp,
                               int vi, int vj)
{
    int gid = blockIdx.x * blockDim.x + threadIdx.x;
    if (gid >= BB * KT * HH * WW) return;
    int b = gid / (KT * HH * WW); int r = gid % (KT * HH * WW);
    int t = r / (HH * WW); int yx = r % (HH * WW);
    int y = yx / WW, xw = yx % WW;
    float du = vu[b] - mg[vi], dv = vv[b] - mg[vj];
    float nrm = sqrtf(du * du + dv * dv + 1e-8f);
    float pu = du / nrm, pv = dv / nrm;
    float sc = scf[((size_t)(b * HH + y) * WW + xw) * 16 + t];
    float py = (float)y  + sc * pu;
    float px = (float)xw + sc * pv;
    float y0f = floorf(py), x0f = floorf(px);
    float wy = py - y0f, wx = px - x0f;
    int y0 = (int)y0f, x0 = (int)x0f, y1 = y0 + 1, x1 = x0 + 1;
    bool vy0 = (y0 >= 0) && (y0 < HH), vy1 = (y1 >= 0) && (y1 < HH);
    bool vx0 = (x0 >= 0) && (x0 < WW), vx1 = (x1 >= 0) && (x1 < WW);
    float w00 = (vy0 && vx0) ? (1.f - wy) * (1.f - wx) : 0.f;
    float w01 = (vy0 && vx1) ? (1.f - wy) * wx         : 0.f;
    float w10 = (vy1 && vx0) ? wy * (1.f - wx)         : 0.f;
    float w11 = (vy1 && vx1) ? wy * wx                 : 0.f;
    int cy0 = min(max(y0, 0), HH - 1), cy1 = min(max(y1, 0), HH - 1);
    int cx0 = min(max(x0, 0), WW - 1), cx1 = min(max(x1, 0), WW - 1);
    int i00 = cy0 * WW + cx0, i01 = cy0 * WW + cx1;
    int i10 = cy1 * WW + cx0, i11 = cy1 * WW + cx1;
    const float* base = x + ((size_t)b * CC * UU * VV + (size_t)vi * VV + vj) * (HH * WW);
    _Float16* dst = samp + ((size_t)(b * HH + y) * WW + xw) * KDEF + t * CC;
#pragma unroll 4
    for (int c = 0; c < CC; ++c) {
        const float* pl = base + (size_t)c * UU * VV * HH * WW;
        float g = w00 * pl[i00] + w01 * pl[i01] + w10 * pl[i10] + w11 * pl[i11];
        dst[c] = (_Float16)g;
    }
}

// A/B fragment loader for v_wmma_f32_16x16x32_f16 (ISA 16-bit A 16x32 layout):
// lanes 0-15: K {0..7,16..23}; lanes 16-31: K {8..15,24..31} of row (lane&15).
DEV v16h mk_frag(const _Float16* rowp, int kc, int half, bool valid)
{
    v8h lo = {}, hi = {};
    if (valid) {
        lo = *(const v8h*)(rowp + kc + half * 8);
        hi = *(const v8h*)(rowp + kc + 16 + half * 8);
    }
    return __builtin_shufflevector(lo, hi, 0, 1, 2, 3, 4, 5, 6, 7,
                                           8, 9, 10, 11, 12, 13, 14, 15);
}

// ---------------- implicit-GEMM conv via WMMA ----------------
// 128 threads = 4 waves = one image row; wave owns 16 pixels x (NT*16) channels.
// Weights double-buffered in LDS via TDM (tensor_load_to_lds) when available.
// OUT_MODE: 0 = f16 NHWC, 1 = f32 NHWC (score conv), 2 = f32 NCHW(UV) into d_out.
template <int CIN, int NT, int TAPS, int OUT_MODE, bool PRELU, bool DEFORM, int COUT_VALID>
__global__ __launch_bounds__(BLOCK)
void conv_wmma_kernel(const _Float16* __restrict__ act,
                      const _Float16* __restrict__ wgt,
                      const float* __restrict__ bias,
                      const float* __restrict__ alpha,
                      _Float16* __restrict__ outh,
                      float* __restrict__ outf,
                      int vi, int vj)
{
    constexpr int COUT    = NT * 16;
    constexpr int ASTRIDE = DEFORM ? KDEF : CIN;
    __shared__ _Float16 wlds[2][CIN * COUT];

    const int tid  = threadIdx.x;
    const int b    = blockIdx.y;
    const int wave = tid >> 5, lane = tid & 31;
    const int row  = blockIdx.x;
    const int px0  = wave * 16;
    const int mrow = lane & 15, half = lane >> 4;

    v8f acc[NT];
#pragma unroll
    for (int nt = 0; nt < NT; ++nt) {
        int oc = nt * 16 + mrow;
        float bb = (oc < COUT_VALID) ? bias[oc] : 0.f;
#pragma unroll
        for (int e = 0; e < 8; ++e) acc[nt][e] = bb;
    }

#if USE_TDM
    constexpr unsigned NDW = (unsigned)(CIN * COUT / 2);   // dwords per tap matrix
    if (wave == 0) {
        tdm_load_1d((unsigned)(size_t)&wlds[0][0], wgt, NDW);
        __builtin_amdgcn_s_wait_tensorcnt(0);
    }
    __syncthreads();
#endif

    for (int t = 0; t < TAPS; ++t) {
        const int cur = t & 1;
#if USE_TDM
        if (t + 1 < TAPS && wave == 0)                     // prefetch next tap via TDM
            tdm_load_1d((unsigned)(size_t)&wlds[cur ^ 1][0],
                        wgt + (size_t)(t + 1) * CIN * COUT, NDW);
#else
        __syncthreads();
        for (int idx = tid * 8; idx < CIN * COUT; idx += BLOCK * 8)
            *(v8h*)&wlds[cur][idx] = *(const v8h*)&wgt[(size_t)t * CIN * COUT + idx];
        __syncthreads();
#endif

        const int dy = (!DEFORM) ? (t / 3 - 1) : 0;
        const int dx = (!DEFORM) ? (t % 3 - 1) : 0;
        const int y  = row + dy;
        const int xm = px0 + mrow + dx;
        const bool valid = DEFORM || ((y >= 0) && (y < HH) && (xm >= 0) && (xm < WW));
        const _Float16* rowp = act
            + ((size_t)(b * HH + (valid ? y : 0)) * WW + (valid ? xm : 0)) * ASTRIDE
            + (DEFORM ? t * CIN : 0);
        __builtin_prefetch(rowp + (size_t)WW * ASTRIDE, 0, 1);  // next row

#pragma unroll
        for (int kc = 0; kc < CIN; kc += 32) {
            v16h af = mk_frag(rowp, kc, half, valid);
#pragma unroll
            for (int nt = 0; nt < NT; ++nt) {
                const _Float16* wrow = &wlds[cur][(nt * 16 + mrow) * CIN];
                v16h bf = mk_frag(wrow, kc, half, true);
                acc[nt] = __builtin_amdgcn_wmma_f32_16x16x32_f16(
                    false, af, false, bf, (short)0, acc[nt], false, false);
            }
        }

#if USE_TDM
        if (t + 1 < TAPS) {
            if (wave == 0) __builtin_amdgcn_s_wait_tensorcnt(0);
            __syncthreads();                               // publish next buffer
        }
#endif
    }

    const float a0 = PRELU ? alpha[0] : 0.f;
#pragma unroll
    for (int nt = 0; nt < NT; ++nt) {
        int oc = nt * 16 + mrow;
#pragma unroll
        for (int e = 0; e < 8; ++e) {
            int m  = e + 8 * half;                 // C/D layout: M = vgpr + 8*(lane>=16)
            int xo = px0 + m;
            float val = acc[nt][e];
            if (PRELU) val = (val >= 0.f) ? val : a0 * val;
            if (OUT_MODE == 0)
                outh[((size_t)(b * HH + row) * WW + xo) * COUT + oc] = (_Float16)val;
            else if (OUT_MODE == 1)
                outf[((size_t)(b * HH + row) * WW + xo) * COUT + oc] = val;
            else
                outf[(((((size_t)b * CC + oc) * UU + vi) * VV + vj) * HH + row) * WW + xo] = val;
        }
    }
}

extern "C" void kernel_launch(void* const* d_in, const int* in_sizes, int n_in,
                              void* d_out, int out_size, void* d_ws, size_t ws_size,
                              hipStream_t stream)
{
    (void)in_sizes; (void)n_in; (void)out_size; (void)ws_size;
    const float* x  = (const float*)d_in[0];
    const float* vu = (const float*)d_in[1];
    const float* vv = (const float*)d_in[2];
    const float* mg = (const float*)d_in[3];
    const float* w1 = (const float*)d_in[4];
    const float* b1 = (const float*)d_in[5];
    const float* a1 = (const float*)d_in[6];
    const float* w2 = (const float*)d_in[7];
    const float* b2 = (const float*)d_in[8];
    const float* a2 = (const float*)d_in[9];
    const float* w3 = (const float*)d_in[10];
    const float* b3 = (const float*)d_in[11];
    const float* a3 = (const float*)d_in[12];
    const float* wf = (const float*)d_in[13];
    const float* bf = (const float*)d_in[14];
    const float* wd = (const float*)d_in[15];
    const float* bd = (const float*)d_in[16];
    const float* a0 = (const float*)d_in[17];
    float* out = (float*)d_out;
    char* ws = (char*)d_ws;

    _Float16* W1g  = (_Float16*)(ws + WS_W1);
    _Float16* W2g  = (_Float16*)(ws + WS_W2);
    _Float16* W3g  = (_Float16*)(ws + WS_W3);
    _Float16* WFg  = (_Float16*)(ws + WS_WF);
    _Float16* WDg  = (_Float16*)(ws + WS_WD);
    _Float16* AIN  = (_Float16*)(ws + WS_AIN);
    _Float16* AH1  = (_Float16*)(ws + WS_AH1);
    _Float16* AH2  = (_Float16*)(ws + WS_AH2);
    _Float16* AH3  = (_Float16*)(ws + WS_AH3);
    float*    SCF  = (float*)(ws + WS_SCF);
    _Float16* SAMP = (_Float16*)(ws + WS_SAMP);

    prep_weights<<<(KT * CC * CIN1 + 255) / 256, 256, 0, stream>>>(
        w1, w2, w3, wf, wd, W1g, W2g, W3g, WFg, WDg);

    const dim3 cgrid(HH, BB);
    for (int vi = 0; vi < UU; ++vi)
        for (int vj = 0; vj < VV; ++vj) {
            pack_input<<<(BB * HH * WW + 255) / 256, 256, 0, stream>>>(
                x, vu, vv, mg, AIN, vi, vj);

            conv_wmma_kernel<CIN1, 4, 9, 0, true,  false, 64><<<cgrid, BLOCK, 0, stream>>>(
                AIN, W1g, b1, a1, AH1, nullptr, vi, vj);
            conv_wmma_kernel<CC,   4, 9, 0, true,  false, 64><<<cgrid, BLOCK, 0, stream>>>(
                AH1, W2g, b2, a2, AH2, nullptr, vi, vj);
            conv_wmma_kernel<CC,   4, 9, 0, true,  false, 64><<<cgrid, BLOCK, 0, stream>>>(
                AH2, W3g, b3, a3, AH3, nullptr, vi, vj);
            conv_wmma_kernel<CC,   1, 9, 1, false, false, 9 ><<<cgrid, BLOCK, 0, stream>>>(
                AH3, WFg, bf, bf, nullptr, SCF, vi, vj);

            gather_samples<<<(BB * KT * HH * WW + 255) / 256, 256, 0, stream>>>(
                x, SCF, vu, vv, mg, SAMP, vi, vj);

            conv_wmma_kernel<CC,   4, 9, 2, true,  true,  64><<<cgrid, BLOCK, 0, stream>>>(
                SAMP, WDg, bd, a0, nullptr, out, vi, vj);
        }
}
